// SymGAT_53180285059706
// MI455X (gfx1250) — matrix operand order repaired and checked by reading, not compiled, not run on hardware
//
#include <hip/hip_runtime.h>
#include <hip/hip_bf16.h>

#define N_NODES 50000
#define N_EDGES 800000
#define DIM 128
#define LN_EPS 1e-5f

typedef __attribute__((ext_vector_type(16))) _Float16 v16h;
typedef __attribute__((ext_vector_type(8)))  _Float16 f16x8;
typedef __attribute__((ext_vector_type(8)))  float    v8f;

// ---------------------------------------------------------------------------
// WMMA helpers: D(16x16,f32) = A(16x32,f16) x B(32x16,f16) + C
// Layouts used here (contiguous k order, same for A and B):
//   A: lane m=lane&15; two contiguous 8-half chunks: K = k0+16*half, k0+16*half+8
//   B: lane n=lane&15; half j -> K = k0 + j + 16*half
//   C: c[r] -> M = r + 8*half, N = lane&15
// Weights are pre-swizzled so each lane's B fragment is one contiguous 32-byte
// v16h load:  Wp[(((k0/32)*128 + n)*2 + half)*16 + j] = W[(k0+j+16*half)*128+n]
// ---------------------------------------------------------------------------
__device__ __forceinline__ v8f wmma_step(v16h a, v16h b, v8f c) {
  return __builtin_amdgcn_wmma_f32_16x16x32_f16(false, a, false, b, (short)0, c,
                                                false, false);
}

__device__ __forceinline__ v16h load_afrag(const _Float16* row, int k0, int half) {
  const f16x8* p = (const f16x8*)row;           // row is 16B-aligned LDS
  f16x8 lo = p[(k0 >> 3) + 2 * half];
  f16x8 hi = p[(k0 >> 3) + 2 * half + 1];
  v16h a;
#pragma unroll
  for (int i = 0; i < 8; ++i) { a[i] = lo[i]; a[8 + i] = hi[i]; }
  return a;
}

__device__ __forceinline__ v16h load_bfrag(const _Float16* __restrict__ Wp,
                                           int k0, int half, int n) {
  const v16h* p = (const v16h*)Wp;              // 32B-aligned packed fragments
  return p[((size_t)(k0 >> 5) * DIM + n) * 2 + half];
}

__device__ __forceinline__ float lrelu(float x) { return x > 0.f ? x : 0.01f * x; }

__device__ __forceinline__ void atomicMaxF(float* addr, float val) {
  int* ia = (int*)addr;
  int cur = __float_as_int(*addr);
  while (__int_as_float(cur) < val) {
    int old = atomicCAS(ia, cur, __float_as_int(val));
    if (old == cur) break;
    cur = old;
  }
}

// ---------------------------------------------------------------------------
// fp32 -> fp16 weight conversion + swizzle into WMMA B-fragment order.
// Source W is [K][128] row-major; i enumerates K*128 elements.
// ---------------------------------------------------------------------------
__global__ void cvt_pack_kernel(const float* __restrict__ src,
                                _Float16* __restrict__ dst, int n) {
  int i = blockIdx.x * blockDim.x + threadIdx.x;
  if (i >= n) return;
  int k = i >> 7, col = i & 127;
  int ktile = k >> 5, rem = k & 31, half = rem >> 4, j = rem & 15;
  dst[(((size_t)ktile * DIM + col) * 2 + half) * 16 + j] = (_Float16)src[i];
}

// Zero agg buffers / denominators; -inf the segment-max buffers.
__global__ void init_ws_kernel(float* __restrict__ aggs, float* __restrict__ dens,
                               float* __restrict__ maxes) {
  size_t i = (size_t)blockIdx.x * blockDim.x + threadIdx.x;
  if (i < (size_t)4 * N_NODES * DIM) aggs[i] = 0.f;
  if (i < (size_t)4 * N_NODES) { dens[i] = 0.f; maxes[i] = -3.0e38f; }
}

// ---------------------------------------------------------------------------
// Node GEMM: Out[M x 128] = A[M x 128] @ W[128 x 128] (+ bias)
// Block = 256 thr (8 waves); block does 16 rows x 128 cols; wave -> 16x16 tile.
// ---------------------------------------------------------------------------
__global__ __launch_bounds__(256) void gemm_node_kernel(
    const float* __restrict__ A, const _Float16* __restrict__ Wp,
    const float* __restrict__ bias, float* __restrict__ Out) {
  __shared__ __align__(16) _Float16 As[16][DIM];
  const int row0 = blockIdx.x * 16;
  const int tid = threadIdx.x;
  {
    int r = tid >> 4, c0 = (tid & 15) * 8;
    const float* sp = A + (size_t)(row0 + r) * DIM + c0;
#pragma unroll
    for (int i = 0; i < 8; ++i) As[r][c0 + i] = (_Float16)sp[i];
  }
  __syncthreads();
  const int wave = tid >> 5, lane = tid & 31, half = lane >> 4, nl = lane & 15;
  const int n = wave * 16 + nl;
  v8f c = {};
#pragma unroll
  for (int k0 = 0; k0 < DIM; k0 += 32)
    c = wmma_step(load_afrag(&As[nl][0], k0, half), load_bfrag(Wp, k0, half, n), c);
  const float bn = bias ? bias[n] : 0.f;
#pragma unroll
  for (int r = 0; r < 8; ++r)
    Out[(size_t)(row0 + r + 8 * half) * DIM + n] = c[r] + bn;
}

// ---------------------------------------------------------------------------
// Fused edge kernel: per 16-edge tile
//   B3  = e @ W_B3                                   (WMMA)
//   eji = relu(B3 + bB3 + B1h[src] + B2h[dst]) + e   (gathers hit L2)
//   z_e = eji @ W_fce                                (WMMA)
//   s2/se2 = z_e . a_attn[128:256] / a_attn_e[128:256]
// ---------------------------------------------------------------------------
__global__ __launch_bounds__(256) void edge_fused_kernel(
    const float* __restrict__ E, const int* __restrict__ src,
    const int* __restrict__ dst, const float* __restrict__ B1h,
    const float* __restrict__ B2h, const float* __restrict__ bB3,
    const _Float16* __restrict__ Wp3, const _Float16* __restrict__ WpFE,
    const float* __restrict__ a2, const float* __restrict__ ae2,
    float* __restrict__ Eji, float* __restrict__ Ze, float* __restrict__ s2,
    float* __restrict__ se2) {
  __shared__ __align__(16) _Float16 Ae[16][DIM];
  __shared__ __align__(16) _Float16 Bji[16][DIM];
  __shared__ float ZeS[16][DIM];
  const int e0 = blockIdx.x * 16;
  const int tid = threadIdx.x;
  {
    int r = tid >> 4, c0 = (tid & 15) * 8;
    const float* sp = E + (size_t)(e0 + r) * DIM + c0;
#pragma unroll
    for (int i = 0; i < 8; ++i) Ae[r][c0 + i] = (_Float16)sp[i];
    if (e0 + 16 < N_EDGES)  // gfx1250 global_prefetch_b8 for the next tile
      __builtin_prefetch(E + (size_t)(e0 + 16 + r) * DIM + c0, 0, 1);
  }
  __syncthreads();
  const int wave = tid >> 5, lane = tid & 31, half = lane >> 4, nl = lane & 15;
  const int n = wave * 16 + nl;
  // GEMM1: B3 = e @ W_B3
  v8f c = {};
#pragma unroll
  for (int k0 = 0; k0 < DIM; k0 += 32)
    c = wmma_step(load_afrag(&Ae[nl][0], k0, half), load_bfrag(Wp3, k0, half, n), c);
  const float bn = bB3[n];
#pragma unroll
  for (int r = 0; r < 8; ++r) {
    const int mr = r + 8 * half, ei = e0 + mr;
    const int s = src[ei], d = dst[ei];
    float v = c[r] + bn + B1h[(size_t)s * DIM + n] + B2h[(size_t)d * DIM + n];
    v = fmaxf(v, 0.f) + E[(size_t)ei * DIM + n];  // exact fp32 residual
    Eji[(size_t)ei * DIM + n] = v;
    Bji[mr][n] = (_Float16)v;
  }
  __syncthreads();
  // GEMM2: z_e = e_ji @ W_fce
  v8f cz = {};
#pragma unroll
  for (int k0 = 0; k0 < DIM; k0 += 32)
    cz = wmma_step(load_afrag(&Bji[nl][0], k0, half), load_bfrag(WpFE, k0, half, n), cz);
#pragma unroll
  for (int r = 0; r < 8; ++r) {
    const int mr = r + 8 * half;
    Ze[(size_t)(e0 + mr) * DIM + n] = cz[r];
    ZeS[mr][n] = cz[r];
  }
  __syncthreads();
  if (tid < 32) {  // per-edge middle attention dot products
    const int row = tid & 15;
    const float* av = (tid < 16) ? a2 : ae2;
    float acc = 0.f;
    for (int k = 0; k < DIM; ++k) acc += ZeS[row][k] * av[k];
    if (tid < 16) s2[e0 + row] = acc; else se2[e0 + row] = acc;
  }
}

// Node-level attention matvecs on z (tiny; z is L2-resident)
__global__ void node_attn_kernel(const float* __restrict__ Z,
                                 const float* __restrict__ a,
                                 const float* __restrict__ ae,
                                 float* __restrict__ zs1, float* __restrict__ zs3,
                                 float* __restrict__ ze1, float* __restrict__ ze3) {
  int nid = blockIdx.x * blockDim.x + threadIdx.x;
  if (nid >= N_NODES) return;
  const float* row = Z + (size_t)nid * DIM;
  float s1 = 0, s3 = 0, e1 = 0, e3 = 0;
  for (int k = 0; k < DIM; ++k) {
    float z = row[k];
    s1 += z * a[k];  s3 += z * a[256 + k];
    e1 += z * ae[k]; e3 += z * ae[256 + k];
  }
  zs1[nid] = s1; zs3[nid] = s3; ze1[nid] = e1; ze3[nid] = e3;
}

// Pass A: 4 leaky-relu scores per edge (fwd/rev x node/edge) + segment max
__global__ void edge_scores_kernel(
    const int* __restrict__ src, const int* __restrict__ dst,
    const float* __restrict__ s2, const float* __restrict__ se2,
    const float* __restrict__ zs1, const float* __restrict__ zs3,
    const float* __restrict__ ze1, const float* __restrict__ ze3,
    float* __restrict__ sf, float* __restrict__ sfe, float* __restrict__ sr,
    float* __restrict__ sre, float* __restrict__ mf, float* __restrict__ mfe,
    float* __restrict__ mr, float* __restrict__ mre) {
  int ei = blockIdx.x * blockDim.x + threadIdx.x;
  if (ei >= N_EDGES) return;
  const int s = src[ei], d = dst[ei];
  const float v2 = s2[ei], v2e = se2[ei];
  float a = lrelu(zs1[s] + v2 + zs3[d]);
  float b = lrelu(ze1[s] + v2e + ze3[d]);
  float c = lrelu(zs1[d] + v2 + zs3[s]);
  float e = lrelu(ze1[d] + v2e + ze3[s]);
  sf[ei] = a; sfe[ei] = b; sr[ei] = c; sre[ei] = e;
  atomicMaxF(mf + d, a);  atomicMaxF(mfe + d, b);
  atomicMaxF(mr + s, c);  atomicMaxF(mre + s, e);
}

// Pass B: ex = exp(score - max[seg]) (in place) + segment sums
__global__ void edge_exp_kernel(
    const int* __restrict__ src, const int* __restrict__ dst,
    float* __restrict__ sf, float* __restrict__ sfe, float* __restrict__ sr,
    float* __restrict__ sre, const float* __restrict__ mf,
    const float* __restrict__ mfe, const float* __restrict__ mr,
    const float* __restrict__ mre, float* __restrict__ df,
    float* __restrict__ dfe, float* __restrict__ dr, float* __restrict__ dre) {
  int ei = blockIdx.x * blockDim.x + threadIdx.x;
  if (ei >= N_EDGES) return;
  const int s = src[ei], d = dst[ei];
  float a = __expf(sf[ei] - mf[d]);  sf[ei] = a;  atomicAdd(df + d, a);
  float b = __expf(sfe[ei] - mfe[d]); sfe[ei] = b; atomicAdd(dfe + d, b);
  float c = __expf(sr[ei] - mr[s]);  sr[ei] = c;  atomicAdd(dr + s, c);
  float e = __expf(sre[ei] - mre[s]); sre[ei] = e; atomicAdd(dre + s, e);
}

// Pass C: scatter-aggregate (128 lanes per edge, atomics into L2-resident aggs)
__global__ __launch_bounds__(256) void edge_agg_kernel(
    const int* __restrict__ src, const int* __restrict__ dst,
    const float* __restrict__ exf, const float* __restrict__ exfe,
    const float* __restrict__ exr, const float* __restrict__ exre,
    const float* __restrict__ df, const float* __restrict__ dfe,
    const float* __restrict__ dr, const float* __restrict__ dre,
    const float* __restrict__ Z, const float* __restrict__ Ze,
    float* __restrict__ aFh, float* __restrict__ aFe, float* __restrict__ aRh,
    float* __restrict__ aRe) {
  size_t gid = (size_t)blockIdx.x * blockDim.x + threadIdx.x;
  const int ei = (int)(gid >> 7), col = (int)(gid & 127);
  if (ei >= N_EDGES) return;
  const int s = src[ei], d = dst[ei];
  const float af = exf[ei] / df[d], afe = exfe[ei] / dfe[d];
  const float ar = exr[ei] / dr[s], are = exre[ei] / dre[s];
  const float zs = Z[(size_t)s * DIM + col];
  const float zd = Z[(size_t)d * DIM + col];
  const float ze = Ze[(size_t)ei * DIM + col];
  atomicAdd(aFh + (size_t)d * DIM + col, af * zs);
  atomicAdd(aFe + (size_t)d * DIM + col, afe * ze);
  atomicAdd(aRh + (size_t)s * DIM + col, ar * zd);
  atomicAdd(aRe + (size_t)s * DIM + col, are * ze);
}

// Mix + LayerNorm + relu + residual.
// h_f + h_r == concat(aFh+aRh, aFe+aRe) @ W_mix + 2*b_mix, single WMMA GEMM K=256.
__global__ __launch_bounds__(256) void mix_ln_kernel(
    const float* __restrict__ aFh, const float* __restrict__ aFe,
    const float* __restrict__ aRh, const float* __restrict__ aRe,
    const _Float16* __restrict__ WpMix, const float* __restrict__ bMix,
    const float* __restrict__ ln_g, const float* __restrict__ ln_b,
    const float* __restrict__ H, float* __restrict__ Out) {
  __shared__ __align__(16) _Float16 As[16][2 * DIM];
  __shared__ float Ho[16][DIM];
  __shared__ float mu[16], rs[16];
  const int row0 = blockIdx.x * 16;
  const int tid = threadIdx.x;
  {
    int r = tid >> 4, c0 = (tid & 15) * 16;
    size_t node = (size_t)(row0 + r);
#pragma unroll
    for (int i = 0; i < 16; ++i) {
      int k = c0 + i;
      float v = (k < DIM) ? (aFh[node * DIM + k] + aRh[node * DIM + k])
                          : (aFe[node * DIM + (k - DIM)] + aRe[node * DIM + (k - DIM)]);
      As[r][k] = (_Float16)v;
    }
  }
  __syncthreads();
  const int wave = tid >> 5, lane = tid & 31, half = lane >> 4, nl = lane & 15;
  const int n = wave * 16 + nl;
  v8f c = {};
#pragma unroll
  for (int k0 = 0; k0 < 2 * DIM; k0 += 32)
    c = wmma_step(load_afrag(&As[nl][0], k0, half), load_bfrag(WpMix, k0, half, n), c);
  const float bn = 2.f * bMix[n];
#pragma unroll
  for (int r = 0; r < 8; ++r) Ho[r + 8 * half][n] = c[r] + bn;
  __syncthreads();
  if (tid < 16) {
    float sum = 0.f, sq = 0.f;
    for (int k = 0; k < DIM; ++k) { float v = Ho[tid][k]; sum += v; sq += v * v; }
    float m = sum * (1.f / DIM);
    mu[tid] = m;
    rs[tid] = rsqrtf(sq * (1.f / DIM) - m * m + LN_EPS);
  }
  __syncthreads();
#pragma unroll
  for (int i = 0; i < 8; ++i) {
    int idx = tid * 8 + i, r = idx >> 7, col = idx & 127;
    float v = (Ho[r][col] - mu[r]) * rs[r] * ln_g[col] + ln_b[col];
    v = fmaxf(v, 0.f) + H[(size_t)(row0 + r) * DIM + col];
    Out[(size_t)(row0 + r) * DIM + col] = v;
  }
}

// ---------------------------------------------------------------------------
extern "C" void kernel_launch(void* const* d_in, const int* in_sizes, int n_in,
                              void* d_out, int out_size, void* d_ws, size_t ws_size,
                              hipStream_t stream) {
  (void)in_sizes; (void)n_in; (void)out_size; (void)ws_size;
  const float* h      = (const float*)d_in[0];
  const float* e      = (const float*)d_in[1];
  const int*   src    = (const int*)d_in[2];
  const int*   dst    = (const int*)d_in[3];
  const float* W_fc   = (const float*)d_in[4];
  const float* W_fce  = (const float*)d_in[5];
  const float* a_attn = (const float*)d_in[6];
  const float* a_attne= (const float*)d_in[7];
  const float* W_B1   = (const float*)d_in[8];
  const float* b_B1   = (const float*)d_in[9];
  const float* W_B2   = (const float*)d_in[10];
  const float* b_B2   = (const float*)d_in[11];
  const float* W_B3   = (const float*)d_in[12];
  const float* b_B3   = (const float*)d_in[13];
  const float* W_mix  = (const float*)d_in[14];
  const float* b_mix  = (const float*)d_in[15];
  const float* ln_g   = (const float*)d_in[16];
  const float* ln_b   = (const float*)d_in[17];

  const size_t NN = N_NODES, NE = N_EDGES;
  float* w = (float*)d_ws;                       // needs ~611 MB of scratch
  float* B1h = w;  w += NN * DIM;
  float* B2h = w;  w += NN * DIM;
  float* Z   = w;  w += NN * DIM;
  float* Ze  = w;  w += NE * DIM;
  float* s2  = w;  w += NE;
  float* se2 = w;  w += NE;
  float* sf  = w;  w += NE;
  float* sfe = w;  w += NE;
  float* sr  = w;  w += NE;
  float* sre = w;  w += NE;
  float* zs1 = w;  w += NN;
  float* zs3 = w;  w += NN;
  float* ze1 = w;  w += NN;
  float* ze3 = w;  w += NN;
  float* maxes = w; w += 4 * NN;                 // mf, mfe, mr, mre
  float* dens  = w; w += 4 * NN;                 // df, dfe, dr, dre
  float* aggs  = w; w += 4 * NN * DIM;           // aFh, aFe, aRh, aRe
  _Float16* wp_fc  = (_Float16*)w;               // packed B-fragment weights
  _Float16* wp_fce = wp_fc + 16384;
  _Float16* wp1    = wp_fce + 16384;
  _Float16* wp2    = wp1 + 16384;
  _Float16* wp3    = wp2 + 16384;
  _Float16* wp_mix = wp3 + 16384;                // 256x128

  float* mf = maxes, *mfe = maxes + NN, *mr = maxes + 2 * NN, *mre = maxes + 3 * NN;
  float* df = dens,  *dfe = dens + NN,  *dr = dens + 2 * NN,  *dre = dens + 3 * NN;
  float* aFh = aggs, *aFe = aggs + NN * DIM, *aRh = aggs + 2 * NN * DIM,
       * aRe = aggs + 3 * NN * DIM;

  float* out_h = (float*)d_out;
  float* out_e = out_h + NN * DIM;               // e_ji section of the output

  // 0) weight conversion + fragment packing + workspace init
  cvt_pack_kernel<<<64, 256, 0, stream>>>(W_fc,  wp_fc,  DIM * DIM);
  cvt_pack_kernel<<<64, 256, 0, stream>>>(W_fce, wp_fce, DIM * DIM);
  cvt_pack_kernel<<<64, 256, 0, stream>>>(W_B1,  wp1,    DIM * DIM);
  cvt_pack_kernel<<<64, 256, 0, stream>>>(W_B2,  wp2,    DIM * DIM);
  cvt_pack_kernel<<<64, 256, 0, stream>>>(W_B3,  wp3,    DIM * DIM);
  cvt_pack_kernel<<<128, 256, 0, stream>>>(W_mix, wp_mix, 2 * DIM * DIM);
  init_ws_kernel<<<(unsigned)(4 * NN * DIM / 256), 256, 0, stream>>>(aggs, dens, maxes);

  // 1) node GEMMs (WMMA)
  gemm_node_kernel<<<N_NODES / 16, 256, 0, stream>>>(h, wp1, b_B1, B1h);
  gemm_node_kernel<<<N_NODES / 16, 256, 0, stream>>>(h, wp2, b_B2, B2h);
  gemm_node_kernel<<<N_NODES / 16, 256, 0, stream>>>(h, wp_fc, nullptr, Z);

  // 2) fused edge update + edge GEMMs (WMMA) + edge attention dots
  edge_fused_kernel<<<N_EDGES / 16, 256, 0, stream>>>(
      e, src, dst, B1h, B2h, b_B3, wp3, wp_fce, a_attn + DIM, a_attne + DIM,
      out_e, Ze, s2, se2);

  // 3) node attention matvecs
  node_attn_kernel<<<(N_NODES + 255) / 256, 256, 0, stream>>>(Z, a_attn, a_attne,
                                                              zs1, zs3, ze1, ze3);
  // 4) segment softmax (max, exp+sum) + weighted scatter aggregation
  edge_scores_kernel<<<(N_EDGES + 255) / 256, 256, 0, stream>>>(
      src, dst, s2, se2, zs1, zs3, ze1, ze3, sf, sfe, sr, sre, mf, mfe, mr, mre);
  edge_exp_kernel<<<(N_EDGES + 255) / 256, 256, 0, stream>>>(
      src, dst, sf, sfe, sr, sre, mf, mfe, mr, mre, df, dfe, dr, dre);
  edge_agg_kernel<<<(unsigned)(NE * DIM / 256), 256, 0, stream>>>(
      src, dst, sf, sfe, sr, sre, df, dfe, dr, dre, Z, Ze, aFh, aFe, aRh, aRe);

  // 5) mix GEMM (WMMA, K=256) + LayerNorm + relu + residual
  mix_ln_kernel<<<N_NODES / 16, 256, 0, stream>>>(aFh, aFe, aRh, aRe, wp_mix,
                                                  b_mix, ln_g, ln_b, h, out_h);
}
